// Head_10660108829382
// MI455X (gfx1250) — matrix operand order repaired and checked by reading, not compiled, not run on hardware
//
#include <hip/hip_runtime.h>

// ---------------------------------------------------------------------------
// Types for CDNA5 WMMA (wave32): 16x16x32 bf16 -> f32
// ---------------------------------------------------------------------------
typedef __attribute__((ext_vector_type(16))) __bf16        v16bf;
typedef __attribute__((ext_vector_type(8)))  float         v8f;
typedef __attribute__((ext_vector_type(8)))  unsigned int  v8u;
typedef __attribute__((ext_vector_type(4)))  unsigned int  v4u;
typedef __attribute__((ext_vector_type(8)))  int           v8i;
typedef __attribute__((ext_vector_type(4)))  int           v4i;

__device__ __forceinline__ v8f wmma_bf16(v16bf a, v16bf b, v8f c) {
  // (neg_a, A, neg_b, B, c_mod, C, reuse_a, reuse_b)
  return __builtin_amdgcn_wmma_f32_16x16x32_bf16(false, a, false, b,
                                                 (short)0, c, false, false);
}

__device__ __forceinline__ unsigned short f32_to_bf16(float f) {
  union { float f; unsigned int u; } cv; cv.f = f;
  unsigned int u = cv.u;
  u += 0x7FFFu + ((u >> 16) & 1u);   // round-to-nearest-even
  return (unsigned short)(u >> 16);
}

// ---------------------------------------------------------------------------
// Projection GEMM: out[r][d] = sum_c A[r][c] * W[c][d]   (fp32 in, bf16 out)
// ---------------------------------------------------------------------------
#define PROJ_BR 128
#define PROJ_BK 64

__global__ __launch_bounds__(256) void proj_kernel(
    const float* __restrict__ A, const float* __restrict__ W,
    unsigned short* __restrict__ out, int R, int C) {
  __shared__ alignas(32) unsigned short Alds[PROJ_BR][PROJ_BK];  // [row][k]
  __shared__ alignas(32) unsigned short Wlds[PROJ_BK][128];      // [k][d]

  const int tid  = threadIdx.x;
  const int lane = tid & 31;
  const int wave = tid >> 5;          // 0..7
  const int half = lane >> 4;         // 0/1
  const int l16  = lane & 15;
  const int rowBase = blockIdx.x * PROJ_BR;

  v8f zero = {};
  v8f acc[8];
#pragma unroll
  for (int i = 0; i < 8; ++i) acc[i] = zero;

  for (int k0 = 0; k0 < C; k0 += PROJ_BK) {
    __syncthreads();
    for (int i = tid; i < PROJ_BR * PROJ_BK; i += 256) {
      int r  = i >> 6;
      int kk = i & 63;
      Alds[r][kk] = f32_to_bf16(A[(size_t)(rowBase + r) * C + (k0 + kk)]);
    }
    for (int i = tid; i < PROJ_BK * 128; i += 256) {
      int kk = i >> 7;
      int d  = i & 127;
      Wlds[kk][d] = f32_to_bf16(W[(size_t)(k0 + kk) * 128 + d]);
    }
    __syncthreads();

#pragma unroll
    for (int kc = 0; kc < PROJ_BK; kc += 32) {
      v8u au;
      int m = wave * 16 + l16;
#pragma unroll
      for (int j = 0; j < 8; ++j) {
        int kk = kc + ((j < 4) ? (2 * j + 8 * half)
                               : (16 + 2 * (j - 4) + 8 * half));
        const unsigned int* p = (const unsigned int*)&Alds[m][kk];
        au[j] = *p;
      }
      v16bf af = __builtin_bit_cast(v16bf, au);
#pragma unroll
      for (int ct = 0; ct < 8; ++ct) {
        v16bf bf;
#pragma unroll
        for (int s = 0; s < 16; ++s) {
          int kk = kc + ((s < 8) ? (8 * half + s) : (16 + 8 * half + (s - 8)));
          bf[s] = __builtin_bit_cast(__bf16, Wlds[kk][ct * 16 + l16]);
        }
        acc[ct] = wmma_bf16(af, bf, acc[ct]);
      }
    }
  }

#pragma unroll
  for (int ct = 0; ct < 8; ++ct) {
#pragma unroll
    for (int r = 0; r < 8; ++r) {
      int row = rowBase + wave * 16 + r + 8 * half;
      out[(size_t)row * 128 + ct * 16 + l16] = f32_to_bf16(acc[ct][r]);
    }
  }
}

// ---------------------------------------------------------------------------
// Attention with post-softmax strict-upper mask.
//   out[n] = sum_{m>n} softmax_row(n)[m] * v[m]   (denominator over ALL m)
// K tile: GLOBAL_LOAD_ASYNC_TO_LDS_B128 (ASYNCcnt) -> LDS transpose.
// V tile: Tensor Data Mover tensor_load_to_lds (TENSORcnt), 2D descriptor.
// ---------------------------------------------------------------------------
#define ATT_BQ 128
#define ATT_BM 32

__global__ __launch_bounds__(256) void attn_kernel(
    const unsigned short* __restrict__ Q,   // [B,N,128] bf16
    const unsigned short* __restrict__ K,   // [B,M,128] bf16
    const unsigned short* __restrict__ V,   // [B,M,128] bf16
    float* __restrict__ Out, int N, int M) {
  __shared__ alignas(32) unsigned short Kraw[ATT_BM][128];   // raw K tile [m][d]
  __shared__ alignas(32) unsigned short Ktl[128][ATT_BM];    // K^T tile [d][m]
  __shared__ alignas(32) unsigned short Vl[ATT_BM][128];     // V tile   [m][d]
  __shared__ alignas(32) unsigned short Pl[8][16][ATT_BM];   // per-wave P

  const int tid  = threadIdx.x;
  const int lane = tid & 31;
  const int wave = tid >> 5;
  const int half = lane >> 4;
  const int l16  = lane & 15;
  const int b    = blockIdx.y;
  const int qBase = blockIdx.x * ATT_BQ + wave * 16;
  const size_t qOff  = ((size_t)b * N + qBase) * 128;
  const size_t kvOff = (size_t)b * M * 128;
  const float scale = 0.08838834764831845f;   // 1/sqrt(128)

  const unsigned ktlLdsBase  = (unsigned)(size_t)&Kraw[0][0];
  const unsigned vLdsBase    = (unsigned)(size_t)&Vl[0][0];

  // Preload Q fragments: 4 chunks of K=32 over d
  v16bf qf[4];
#pragma unroll
  for (int c = 0; c < 4; ++c) {
    v8u au;
#pragma unroll
    for (int j = 0; j < 8; ++j) {
      int kk = c * 32 + ((j < 4) ? (2 * j + 8 * half)
                                 : (16 + 2 * (j - 4) + 8 * half));
      const unsigned int* p = (const unsigned int*)&Q[qOff + (size_t)l16 * 128 + kk];
      au[j] = *p;
    }
    qf[c] = __builtin_bit_cast(v16bf, au);
  }

  v8f zero = {};
  v8f o[8];
#pragma unroll
  for (int i = 0; i < 8; ++i) o[i] = zero;
  float rm[8], rl[8];
#pragma unroll
  for (int r = 0; r < 8; ++r) { rm[r] = -1e30f; rl[r] = 0.0f; }

  for (int m0 = 0; m0 < M; m0 += ATT_BM) {
    __syncthreads();   // everyone done reading previous Ktl/Vl/Kraw

    // ---- K tile: async global->LDS copy (8 KB contiguous), ASYNCcnt path ----
    {
      const unsigned long long kTile =
          (unsigned long long)(size_t)&K[kvOff + (size_t)m0 * 128];
#pragma unroll
      for (int i = 0; i < 2; ++i) {
        unsigned off = (unsigned)(wave * 1024 + i * 512 + lane * 16);
        unsigned long long ga = kTile + off;
        unsigned la = ktlLdsBase + off;
        asm volatile("global_load_async_to_lds_b128 %0, %1, off"
                     :: "v"(la), "v"(ga) : "memory");
      }
    }

    // ---- V tile: Tensor Data Mover (one DMA per workgroup), TENSORcnt path ----
    if (wave == 0) {
      const unsigned long long vTile =
          (unsigned long long)(size_t)&V[kvOff + (size_t)m0 * 128];
      v4u g0;
      g0[0] = 1u;                                   // count=1, user descriptor
      g0[1] = vLdsBase;                             // lds_addr
      g0[2] = (unsigned)(vTile & 0xFFFFFFFFull);    // global_addr[31:0]
      g0[3] = (unsigned)((vTile >> 32) & 0x1FFFFFFu) | (2u << 30); // addr[56:32], type=2
      v8i g1;
      g1[0] = (int)(1u << 16);                      // workgroup_mask=0, data_size=1 (2B)
      g1[1] = (int)(128u << 16);                    // tensor_dim0 = 128 (low 16 @bit48)
      g1[2] = (int)((unsigned)M << 16);             // tensor_dim1 (low 16 @bit80)
      g1[3] = (int)(128u << 16);                    // tile_dim0 = 128 (@bit112)
      g1[4] = ATT_BM;                               // tile_dim1 = 32, tile_dim2 = 0
      g1[5] = 128;                                  // tensor_dim0_stride = 128
      g1[6] = 0;                                    // dim1_stride (unused for 2D)
      g1[7] = 0;
      v4i gz4 = {};
      v8i gz8 = {};
      __builtin_amdgcn_tensor_load_to_lds(g0, g1, gz4, gz4, gz8, 0);
    }

    // Wait for both async paths, then make LDS visible to all waves
    asm volatile("s_wait_asynccnt 0" ::: "memory");
    __builtin_amdgcn_s_wait_tensorcnt(0);
    __syncthreads();

    // ---- transpose Kraw[m][d] -> Ktl[d][m] (LDS-resident) ----
    {
      int mm = tid & 31;
      int d0 = (tid >> 5) * 16;
      const unsigned short* kp = &Kraw[mm][d0];
#pragma unroll
      for (int i = 0; i < 16; ++i) Ktl[d0 + i][mm] = kp[i];
      if (m0 + ATT_BM < M) {   // global_prefetch_b8 for next tile
        __builtin_prefetch(&K[kvOff + (size_t)(m0 + ATT_BM + mm) * 128 + d0], 0, 0);
        __builtin_prefetch(&V[kvOff + (size_t)(m0 + ATT_BM + mm) * 128 + d0], 0, 0);
      }
    }
    __syncthreads();

    // ---- S = (Q K^T) * scale for 32 cols (two 16-col WMMA tiles) ----
    v8f s0 = zero, s1 = zero;
#pragma unroll
    for (int c = 0; c < 4; ++c) {
      v16bf b0, b1;
#pragma unroll
      for (int s = 0; s < 16; ++s) {
        int kk = c * 32 + ((s < 8) ? (8 * half + s) : (16 + 8 * half + (s - 8)));
        b0[s] = __builtin_bit_cast(__bf16, Kraw[l16][kk]);
        b1[s] = __builtin_bit_cast(__bf16, Kraw[16 + l16][kk]);
      }
      s0 = wmma_bf16(qf[c], b0, s0);
      s1 = wmma_bf16(qf[c], b1, s1);
    }
#pragma unroll
    for (int r = 0; r < 8; ++r) { s0[r] *= scale; s1[r] *= scale; }

    // ---- online softmax: row max across the 16 lanes holding each row ----
    float sf[8];
#pragma unroll
    for (int r = 0; r < 8; ++r) {
      float v = fmaxf(s0[r], s1[r]);
#pragma unroll
      for (int off = 1; off < 16; off <<= 1)
        v = fmaxf(v, __shfl_xor(v, off, 32));
      float nm = fmaxf(rm[r], v);
      sf[r] = __expf(rm[r] - nm);
      rm[r] = nm;
    }
#pragma unroll
    for (int c = 0; c < 8; ++c)
#pragma unroll
      for (int r = 0; r < 8; ++r) o[c][r] *= sf[r];

    // e = exp(s - max): denominator over all m, numerator masked to m > q
#pragma unroll
    for (int r = 0; r < 8; ++r) {
      int qg = qBase + r + 8 * half;
      float e0 = __expf(s0[r] - rm[r]);
      float e1 = __expf(s1[r] - rm[r]);
      float rs = e0 + e1;
#pragma unroll
      for (int off = 1; off < 16; off <<= 1)
        rs += __shfl_xor(rs, off, 32);
      rl[r] = rl[r] * sf[r] + rs;
      int mg0 = m0 + l16;
      int mg1 = m0 + 16 + l16;
      Pl[wave][r + 8 * half][l16]      = f32_to_bf16(mg0 > qg ? e0 : 0.0f);
      Pl[wave][r + 8 * half][16 + l16] = f32_to_bf16(mg1 > qg ? e1 : 0.0f);
    }
    __builtin_amdgcn_wave_barrier();   // same-wave LDS ordering (DS in-order)

    // Rebuild P as 16x32 A-fragment from per-wave LDS scratch
    v8u pu;
#pragma unroll
    for (int j = 0; j < 8; ++j) {
      int kk = (j < 4) ? (2 * j + 8 * half) : (16 + 2 * (j - 4) + 8 * half);
      const unsigned int* pp = (const unsigned int*)&Pl[wave][l16][kk];
      pu[j] = *pp;
    }
    v16bf pf = __builtin_bit_cast(v16bf, pu);

    // O += P * V : B-fragment lane L holds V column d=c*16+L%16, m's in table order
#pragma unroll
    for (int c = 0; c < 8; ++c) {
      v16bf bv;
#pragma unroll
      for (int s = 0; s < 16; ++s) {
        int kk = (s < 8) ? (8 * half + s) : (16 + 8 * half + (s - 8));
        bv[s] = __builtin_bit_cast(__bf16, Vl[kk][c * 16 + l16]);
      }
      o[c] = wmma_bf16(pf, bv, o[c]);
    }
  }

  // Normalize by full-row softmax denominator and store fp32
#pragma unroll
  for (int c = 0; c < 8; ++c) {
#pragma unroll
    for (int r = 0; r < 8; ++r) {
      int row = qBase + r + 8 * half;
      Out[((size_t)b * N + row) * 128 + c * 16 + l16] = o[c][r] / rl[r];
    }
  }
}

// ---------------------------------------------------------------------------
// Launch
// ---------------------------------------------------------------------------
extern "C" void kernel_launch(void* const* d_in, const int* in_sizes, int n_in,
                              void* d_out, int out_size, void* d_ws, size_t ws_size,
                              hipStream_t stream) {
  const float* x  = (const float*)d_in[0];
  const float* y  = (const float*)d_in[1];
  const float* Wq = (const float*)d_in[2];
  const float* Wk = (const float*)d_in[3];
  const float* Wv = (const float*)d_in[4];
  float* out = (float*)d_out;

  const int B = 4, N = 4096, M = 4096, C = 1024, D = 128;

  unsigned short* q = (unsigned short*)d_ws;                 // [B*N*D] bf16
  unsigned short* k = q + (size_t)B * N * D;                 // [B*M*D] bf16
  unsigned short* v = k + (size_t)B * M * D;                 // [B*M*D] bf16

  const int Rq = B * N, Rk = B * M;
  proj_kernel<<<Rq / PROJ_BR, 256, 0, stream>>>(x, Wq, q, Rq, C);
  proj_kernel<<<Rk / PROJ_BR, 256, 0, stream>>>(y, Wk, k, Rk, C);
  proj_kernel<<<Rk / PROJ_BR, 256, 0, stream>>>(y, Wv, v, Rk, C);

  dim3 grid(N / ATT_BQ, B);
  attn_kernel<<<grid, 256, 0, stream>>>(q, k, v, out, N, M);
}
